// CausalSelfAttention_6554120093801
// MI455X (gfx1250) — matrix operand order repaired and checked by reading, not compiled, not run on hardware
//
#include <hip/hip_runtime.h>

typedef __attribute__((ext_vector_type(16))) __bf16 v16bf;
typedef __attribute__((ext_vector_type(8)))  float  v8f;

// ---------------------------------------------------------------------------
// helpers
// ---------------------------------------------------------------------------
static __device__ __forceinline__ unsigned short f2bf(float f) {
    unsigned int u = __float_as_uint(f);
    u += 0x7FFFu + ((u >> 16) & 1u);          // round-to-nearest-even
    return (unsigned short)(u >> 16);
}

// packed f32x2 -> bf16x2 (hardware v_cvt_pk_bf16_f32 when available)
static __device__ __forceinline__ unsigned int pk2bf(float a, float b) {
#if defined(__has_builtin) && __has_builtin(__builtin_amdgcn_cvt_pk_bf16_f32)
    typedef __attribute__((ext_vector_type(2))) __bf16 v2bf;
    v2bf r = __builtin_amdgcn_cvt_pk_bf16_f32(a, b);
    return __builtin_bit_cast(unsigned int, r);
#else
    return (unsigned int)f2bf(a) | ((unsigned int)f2bf(b) << 16);
#endif
}

// CDNA5 async global->LDS copy (ASYNCcnt-tracked), GVS addressing:
// mem_addr = SGPR64 + VGPR32 ; LDS dest address taken from a VGPR.
#define ASYNC_LDS_B128(lds32, voff32, sbase64)                                 \
    asm volatile("global_load_async_to_lds_b128 %0, %1, %2"                    \
                 :: "v"(lds32), "v"(voff32), "s"(sbase64) : "memory")
#define WAIT_ASYNC() asm volatile("s_wait_asynccnt 0" ::: "memory")

static __device__ __forceinline__ unsigned int lds_addr32(const void* p) {
    return (unsigned int)(size_t)p;            // generic LDS addr low 32 bits
}

union FragU { struct { uint4 lo, hi; } q; v16bf v; };

// A-matrix fragment (16x32 bf16): lane L: M = L%16, half = L/16,
// elements 0..7  -> K = half*8 + 0..7 ; elements 8..15 -> K = 16+half*8+0..7
static __device__ __forceinline__ v16bf load_a_frag(const unsigned short* lds,
                                                    int row0, int ldk, int k0) {
    const int lane = threadIdx.x & 31;
    const int half = lane >> 4;
    const int m    = lane & 15;
    const unsigned short* p = lds + (size_t)(row0 + m) * ldk + k0 + half * 8;
    FragU u;
    u.q.lo = *(const uint4*)(p);
    u.q.hi = *(const uint4*)(p + 16);
    return u.v;
}

// B-matrix fragment (32x16 bf16): lane L: N = L%16, half = L/16,
// elements 0..7  -> K = half*8 + 0..7 ; elements 8..15 -> K = 16+half*8+0..7
static __device__ __forceinline__ v16bf load_b_frag(const unsigned short* lds,
                                                    int col0, int ldk, int k0) {
    const int lane = threadIdx.x & 31;
    const int half = lane >> 4;
    const int n    = lane & 15;
    const unsigned short* p = lds + (size_t)(col0 + n) * ldk + k0 + half * 8;
    FragU u;
    u.q.lo = *(const uint4*)(p);
    u.q.hi = *(const uint4*)(p + 16);
    return u.v;
}

static __device__ __forceinline__ v8f wmma_bf16(v16bf a, v16bf b, v8f c) {
    return __builtin_amdgcn_wmma_f32_16x16x32_bf16(false, a, false, b,
                                                   (short)0, c, false, false);
}

#define Bdim  4
#define Ndim  2048
#define Ddim  1024
#define Hdim  16
#define DHdim 64

#define BM 128
#define BN 128
#define BK 32

// stage a 128x32 fp32 tile -> bf16 LDS tile [r][k] (256 threads)
static __device__ __forceinline__ void stage_f32_tile(const float* __restrict__ src,
                                                      int ld, unsigned short* dst,
                                                      int t) {
    for (int i = 0; i < 4; i++) {
        int f4 = t + i * 256;
        int r = f4 >> 3, c = (f4 & 7) << 2;
        float4 v = *(const float4*)(src + (size_t)r * ld + c);
        *(uint2*)(&dst[r * BK + c]) = make_uint2(pk2bf(v.x, v.y), pk2bf(v.z, v.w));
    }
}

// ---------------------------------------------------------------------------
// Kernel 1: qkv = x @ W_qkv^T, scattered into bf16 Q/K/V [B,H,N,DH]
// double-buffered LDS pipeline, one barrier per K-step
// ---------------------------------------------------------------------------
__global__ __launch_bounds__(256) void qkv_gemm_kernel(
    const float* __restrict__ x,        // [B*N, D]
    const float* __restrict__ Wqkv,     // [3D, D]
    unsigned short* __restrict__ qg,
    unsigned short* __restrict__ kg,
    unsigned short* __restrict__ vg)
{
    __shared__ unsigned short lA[2][BM * BK];
    __shared__ unsigned short lB[2][BN * BK];

    const int t    = threadIdx.x;
    const int wave = t >> 5, lane = t & 31;
    const int half = lane >> 4, ln = lane & 15;
    const int row0 = blockIdx.x * BM;
    const int col0 = blockIdx.y * BN;
    const int wm = wave >> 2;
    const int wn = wave & 3;

    v8f acc[4][2];
    const v8f vzero = {0.f,0.f,0.f,0.f,0.f,0.f,0.f,0.f};
    for (int i = 0; i < 4; i++)
        for (int j = 0; j < 2; j++) acc[i][j] = vzero;

    const float* aBase = x    + (size_t)row0 * Ddim;
    const float* bBase = Wqkv + (size_t)col0 * Ddim;

    stage_f32_tile(aBase, Ddim, lA[0], t);
    stage_f32_tile(bBase, Ddim, lB[0], t);
    __syncthreads();

    int p = 0;
    for (int k0 = 0; k0 < Ddim; k0 += BK) {
        if (k0 + BK < Ddim) {                    // prefetch next tiles
            stage_f32_tile(aBase + k0 + BK, Ddim, lA[p ^ 1], t);
            stage_f32_tile(bBase + k0 + BK, Ddim, lB[p ^ 1], t);
        }
        v16bf bfr0 = load_b_frag(lB[p], wn * 32 +  0, BK, 0);
        v16bf bfr1 = load_b_frag(lB[p], wn * 32 + 16, BK, 0);
        for (int i = 0; i < 4; i++) {
            v16bf afr = load_a_frag(lA[p], wm * 64 + i * 16, BK, 0);
            acc[i][0] = wmma_bf16(afr, bfr0, acc[i][0]);
            acc[i][1] = wmma_bf16(afr, bfr1, acc[i][1]);
        }
        __syncthreads();
        p ^= 1;
    }

    // scatter epilogue: C[m,o] -> {q|k|v}[b,h,n,dh] bf16
    for (int i = 0; i < 4; i++) {
        for (int j = 0; j < 2; j++) {
            int mbase = row0 + wm * 64 + i * 16 + half * 8;
            int o = col0 + wn * 32 + j * 16 + ln;
            int part = o >> 10;
            int d    = o & (Ddim - 1);
            int h    = d >> 6;
            int dh   = d & (DHdim - 1);
            unsigned short* dst = (part == 0) ? qg : ((part == 1) ? kg : vg);
            for (int r = 0; r < 8; r++) {
                int m = mbase + r;
                int b = m >> 11;
                int n = m & (Ndim - 1);
                dst[(((size_t)(b * Hdim + h)) * Ndim + n) * DHdim + dh] =
                    f2bf(acc[i][j][r]);
            }
        }
    }
}

// ---------------------------------------------------------------------------
// Kernel 2: flash attention per (b,h), 64-query block, 4 waves x 16 rows
// K tiles streamed with async global->LDS copies, double-buffered
// ---------------------------------------------------------------------------
#define BR 64
#define BC 32

__global__ __launch_bounds__(128) void attn_kernel(
    const unsigned short* __restrict__ qg,  // [B,H,N,DH] bf16
    const unsigned short* __restrict__ kg,
    const unsigned short* __restrict__ vg,
    unsigned short* __restrict__ yg)        // [B,N,D] bf16
{
    __shared__ unsigned short lQ[BR * DHdim];
    __shared__ unsigned short lK[2][BC * DHdim];
    __shared__ unsigned short lVt[2][DHdim * BC];
    __shared__ unsigned short lP[4][16 * BC];

    const int t    = threadIdx.x;
    const int wave = t >> 5, lane = t & 31;
    const int half = lane >> 4, ln = lane & 15;
    const int qbase = blockIdx.x * BR;
    const int bh = blockIdx.y;
    const int b = bh >> 4, h = bh & (Hdim - 1);
    const size_t headoff = (size_t)bh * Ndim * DHdim;

    const unsigned long long qB = (unsigned long long)(size_t)(qg + headoff + (size_t)qbase * DHdim);
    const unsigned long long kB = (unsigned long long)(size_t)(kg + headoff);
    const unsigned short*    vP = vg + headoff;

    // async stage Q block (64x64 bf16): 128 thr x 2 x b128
    {
        int r0 = t >> 2, c0 = (t & 3) << 4;
        ASYNC_LDS_B128(lds_addr32(&lQ[r0 * DHdim + c0]),
                       (unsigned int)((r0 * DHdim + c0) * 2), qB);
        int r1 = (t + 128) >> 2, c1 = ((t + 128) & 3) << 4;
        ASYNC_LDS_B128(lds_addr32(&lQ[r1 * DHdim + c1]),
                       (unsigned int)((r1 * DHdim + c1) * 2), qB);
    }
    // async stage K block kb=0 (32x64 bf16): 128 thr x 1 x b128
    {
        int r = t >> 2, c = (t & 3) << 4;
        ASYNC_LDS_B128(lds_addr32(&lK[0][r * DHdim + c]),
                       (unsigned int)((r * DHdim + c) * 2), kB);
    }
    // manual stage V^T block kb=0 (transpose: [dh][key])
    for (int i = 0; i < 4; i++) {
        int e = t + i * 128;
        int key = e >> 4, c = (e & 15) << 2;
        uint2 vv = *(const uint2*)(vP + (size_t)key * DHdim + c);
        const unsigned short* pv = (const unsigned short*)&vv;
        lVt[0][(c + 0) * BC + key] = pv[0];
        lVt[0][(c + 1) * BC + key] = pv[1];
        lVt[0][(c + 2) * BC + key] = pv[2];
        lVt[0][(c + 3) * BC + key] = pv[3];
    }
    WAIT_ASYNC();
    __syncthreads();

    v16bf qf0 = load_a_frag(lQ, wave * 16, DHdim, 0);
    v16bf qf1 = load_a_frag(lQ, wave * 16, DHdim, 32);

    const v8f vzero = {0.f,0.f,0.f,0.f,0.f,0.f,0.f,0.f};
    float mrow[8], lrow[8];
    v8f o[4];
    for (int r = 0; r < 8; r++) { mrow[r] = -1e30f; lrow[r] = 0.f; }
    for (int j = 0; j < 4; j++) o[j] = vzero;
    const float scale = 0.125f;   // 1/sqrt(64)

    const int kbmax = (qbase + BR - 1) / BC;
    int p = 0;
    for (int kb = 0; kb <= kbmax; kb++) {
        const int kbase = kb * BC;
        if (kb < kbmax) {                        // prefetch next K/V block
            const int nbase = kbase + BC;
            int r = t >> 2, c = (t & 3) << 4;
            ASYNC_LDS_B128(lds_addr32(&lK[p ^ 1][r * DHdim + c]),
                           (unsigned int)(((nbase + r) * DHdim + c) * 2), kB);
            for (int i = 0; i < 4; i++) {
                int e = t + i * 128;
                int key = e >> 4, cc = (e & 15) << 2;
                uint2 vv = *(const uint2*)(vP + (size_t)(nbase + key) * DHdim + cc);
                const unsigned short* pv = (const unsigned short*)&vv;
                lVt[p ^ 1][(cc + 0) * BC + key] = pv[0];
                lVt[p ^ 1][(cc + 1) * BC + key] = pv[1];
                lVt[p ^ 1][(cc + 2) * BC + key] = pv[2];
                lVt[p ^ 1][(cc + 3) * BC + key] = pv[3];
            }
        }

        // S = Q K^T  (DH=64 -> two chained k=32 WMMAs per 16-key tile)
        v8f s[2];
        s[0] = vzero; s[1] = vzero;
        for (int tt = 0; tt < 2; tt++) {
            v16bf kf0 = load_b_frag(lK[p], tt * 16, DHdim, 0);
            v16bf kf1 = load_b_frag(lK[p], tt * 16, DHdim, 32);
            s[tt] = wmma_bf16(qf0, kf0, s[tt]);
            s[tt] = wmma_bf16(qf1, kf1, s[tt]);
        }

        // scale + causal mask
        const int qrow0 = qbase + wave * 16 + half * 8;
        for (int tt = 0; tt < 2; tt++) {
            int kidx = kbase + tt * 16 + ln;
            for (int r = 0; r < 8; r++) {
                float sv = s[tt][r] * scale;
                if (kidx > qrow0 + r) sv = -1e30f;
                s[tt][r] = sv;
            }
        }

        // online softmax (row reductions stay within a 16-lane half)
        for (int r = 0; r < 8; r++) {
            float mx = fmaxf(s[0][r], s[1][r]);
            for (int d = 1; d < 16; d <<= 1) mx = fmaxf(mx, __shfl_xor(mx, d, 32));
            float mnew  = fmaxf(mrow[r], mx);
            float alpha = __expf(mrow[r] - mnew);
            float p0 = __expf(s[0][r] - mnew);
            float p1 = __expf(s[1][r] - mnew);
            float rs = p0 + p1;
            for (int d = 1; d < 16; d <<= 1) rs += __shfl_xor(rs, d, 32);
            lrow[r] = lrow[r] * alpha + rs;
            mrow[r] = mnew;
            for (int j = 0; j < 4; j++) o[j][r] *= alpha;
            s[0][r] = p0; s[1][r] = p1;
        }

        // P (C-layout) -> per-wave LDS -> A-layout fragment; O += P V
        unsigned short* myP = lP[wave];
        for (int tt = 0; tt < 2; tt++)
            for (int r = 0; r < 8; r++)
                myP[(half * 8 + r) * BC + tt * 16 + ln] = f2bf(s[tt][r]);
        v16bf pf = load_a_frag(myP, 0, BC, 0);   // DS in-order per wave
        for (int j = 0; j < 4; j++) {
            v16bf vf = load_b_frag(lVt[p], j * 16, BC, 0);
            o[j] = wmma_bf16(pf, vf, o[j]);
        }

        WAIT_ASYNC();
        __syncthreads();
        p ^= 1;
    }

    // normalize and write y as bf16 [B,N,D]
    for (int r = 0; r < 8; r++) {
        float inv = 1.0f / lrow[r];
        int qrow = qbase + wave * 16 + half * 8 + r;
        size_t base = ((size_t)(b * Ndim + qrow)) * Ddim + h * DHdim;
        for (int j = 0; j < 4; j++)
            yg[base + j * 16 + ln] = f2bf(o[j][r] * inv);
    }
}

// ---------------------------------------------------------------------------
// Kernel 3: out = y @ W_out^T, fp32 result; A tiles via async LDS copy
// ---------------------------------------------------------------------------
__global__ __launch_bounds__(256) void out_gemm_kernel(
    const unsigned short* __restrict__ y,   // [B*N, D] bf16
    const float* __restrict__ Wout,         // [D, D]
    float* __restrict__ out)                // [B*N, D] fp32
{
    __shared__ unsigned short lA[2][BM * BK];
    __shared__ unsigned short lB[2][BN * BK];

    const int t    = threadIdx.x;
    const int wave = t >> 5, lane = t & 31;
    const int half = lane >> 4, ln = lane & 15;
    const int row0 = blockIdx.x * BM;
    const int col0 = blockIdx.y * BN;
    const int wm = wave >> 2;
    const int wn = wave & 3;

    v8f acc[4][2];
    const v8f vzero = {0.f,0.f,0.f,0.f,0.f,0.f,0.f,0.f};
    for (int i = 0; i < 4; i++)
        for (int j = 0; j < 2; j++) acc[i][j] = vzero;

    const unsigned long long aB = (unsigned long long)(size_t)(y + (size_t)row0 * Ddim);
    const float* bBase = Wout + (size_t)col0 * Ddim;

    // async stage of A tile: 128x32 bf16 = 256 thr x 1 x b128
    const int ar = t >> 1, ac = (t & 1) << 4;
    ASYNC_LDS_B128(lds_addr32(&lA[0][ar * BK + ac]),
                   (unsigned int)((ar * Ddim + ac) * 2), aB);
    stage_f32_tile(bBase, Ddim, lB[0], t);
    WAIT_ASYNC();
    __syncthreads();

    int p = 0;
    for (int k0 = 0; k0 < Ddim; k0 += BK) {
        if (k0 + BK < Ddim) {
            ASYNC_LDS_B128(lds_addr32(&lA[p ^ 1][ar * BK + ac]),
                           (unsigned int)((ar * Ddim + k0 + BK + ac) * 2), aB);
            stage_f32_tile(bBase + k0 + BK, Ddim, lB[p ^ 1], t);
        }
        v16bf bfr0 = load_b_frag(lB[p], wn * 32 +  0, BK, 0);
        v16bf bfr1 = load_b_frag(lB[p], wn * 32 + 16, BK, 0);
        for (int i = 0; i < 4; i++) {
            v16bf afr = load_a_frag(lA[p], wm * 64 + i * 16, BK, 0);
            acc[i][0] = wmma_bf16(afr, bfr0, acc[i][0]);
            acc[i][1] = wmma_bf16(afr, bfr1, acc[i][1]);
        }
        WAIT_ASYNC();
        __syncthreads();
        p ^= 1;
    }

    for (int i = 0; i < 4; i++) {
        for (int j = 0; j < 2; j++) {
            int mbase = row0 + wm * 64 + i * 16 + half * 8;
            int ocol  = col0 + wn * 32 + j * 16 + ln;
            for (int r = 0; r < 8; r++)
                out[(size_t)(mbase + r) * Ddim + ocol] = acc[i][j][r];
        }
    }
}

// ---------------------------------------------------------------------------
// launch
// ---------------------------------------------------------------------------
extern "C" void kernel_launch(void* const* d_in, const int* in_sizes, int n_in,
                              void* d_out, int out_size, void* d_ws, size_t ws_size,
                              hipStream_t stream)
{
    const float* x    = (const float*)d_in[0];
    const float* Wqkv = (const float*)d_in[1];
    const float* Wout = (const float*)d_in[2];
    float* out = (float*)d_out;

    const size_t headElems = (size_t)Bdim * Hdim * Ndim * DHdim; // 8,388,608
    unsigned short* qg = (unsigned short*)d_ws;
    unsigned short* kg = qg + headElems;
    unsigned short* vg = kg + headElems;
    unsigned short* yg = vg + headElems;

    dim3 g1((Bdim * Ndim) / BM, (3 * Ddim) / BN);   // 64 x 24
    qkv_gemm_kernel<<<g1, dim3(256), 0, stream>>>(x, Wqkv, qg, kg, vg);

    dim3 g2(Ndim / BR, Bdim * Hdim);                // 32 x 64
    attn_kernel<<<g2, dim3(128), 0, stream>>>(qg, kg, vg, yg);

    dim3 g3((Bdim * Ndim) / BM, Ddim / BN);         // 64 x 8
    out_gemm_kernel<<<g3, dim3(256), 0, stream>>>(yg, Wout, out);
}
